// GraphConvolution_67044439491107
// MI455X (gfx1250) — compile-verified, matrix operand order checked by reading, not run on hardware
//
#include <hip/hip_runtime.h>
#include <hip/hip_bf16.h>

#define F_IN  256
#define F_OUT 128

typedef __attribute__((ext_vector_type(16))) __bf16    v16bf;
typedef __attribute__((ext_vector_type(8)))  float     v8f;
typedef __attribute__((ext_vector_type(8)))  unsigned  v8u;

// Pack two f32 into one dword of two bf16 (truncation) with a single v_perm_b32.
__device__ __forceinline__ unsigned pack_bf16_pair(float lo, float hi) {
    return __builtin_amdgcn_perm(__float_as_uint(hi), __float_as_uint(lo), 0x07060302u);
}

// Build a 16x32 bf16 A-fragment for one K-step from a f32 row.
// Element e holds K = kk_i*32 + 8*g + e + (e>=8 ? 8 : 0)  (ISA 16-bit A layout).
__device__ __forceinline__ v16bf make_afrag(const float* __restrict__ xrow,
                                            int kk_i, int g) {
    const float4* xp = (const float4*)(xrow + kk_i * 32 + 8 * g);
    float4 p0 = xp[0], p1 = xp[1];       // elements 0..7
    float4 q0 = xp[4], q1 = xp[5];       // elements 8..15
    v8u au;
    au[0] = pack_bf16_pair(p0.x, p0.y);
    au[1] = pack_bf16_pair(p0.z, p0.w);
    au[2] = pack_bf16_pair(p1.x, p1.y);
    au[3] = pack_bf16_pair(p1.z, p1.w);
    au[4] = pack_bf16_pair(q0.x, q0.y);
    au[5] = pack_bf16_pair(q0.z, q0.w);
    au[6] = pack_bf16_pair(q1.x, q1.y);
    au[7] = pack_bf16_pair(q1.z, q1.w);
    v16bf a;
    __builtin_memcpy(&a, &au, sizeof(a));
    return a;
}

// out[n, f] = bias[f]
__global__ __launch_bounds__(256) void init_bias_kernel(
    const float* __restrict__ bias, float* __restrict__ out, int total) {
    int i = blockIdx.x * 256 + threadIdx.x;
    if (i < total) out[i] = bias[i & (F_OUT - 1)];
}

// Pre-pack W[256,128] f32 into per-(kstep, coltile, lane) contiguous bf16 B-fragments.
// Bpack index (v8u units): (kk_i*8 + colT)*32 + lane.
__global__ __launch_bounds__(256) void pack_w_kernel(
    const float* __restrict__ W, unsigned* __restrict__ Bpack) {
    int t = blockIdx.x * 256 + threadIdx.x;      // 0 .. 2047
    if (t >= 8 * 8 * 32) return;
    int lane  = t & 31;
    int n     = ((t >> 5) & 7) * 16 + (lane & 15);
    int kbase = (t >> 8) * 32 + 8 * (lane >> 4);
    v8u frag;
    #pragma unroll
    for (int d = 0; d < 8; ++d) {
        int k0 = kbase + 2 * d + (d >= 4 ? 8 : 0);
        float lo = W[(size_t)k0 * F_OUT + n];
        float hi = W[(size_t)(k0 + 1) * F_OUT + n];
        frag[d] = pack_bf16_pair(lo, hi);
    }
    *((v8u*)Bpack + t) = frag;
}

// support[N,128] = X[N,256] @ W[256,128]; bf16 WMMA, f32 accumulate.
// Block = 256 threads = 8 wave32. Wave w handles cols [16w,16w+16) for TWO
// row tiles: [32*blk, +16) and [32*blk+16, +16). The two WMMAs per K-step are
// independent -> they fill each other's bf16 WMMA hazard/co-exec slots, and
// the B fragment is loaded once for both.
__global__ __launch_bounds__(256) void gemm_xw_kernel(
    const float* __restrict__ X, const unsigned* __restrict__ Bpack,
    float* __restrict__ S, int n_nodes) {
    const int lane = threadIdx.x & 31;
    const int wave = threadIdx.x >> 5;   // col tile 0..7
    const int m    = lane & 15;
    const int g    = lane >> 4;
    const int blk  = blockIdx.x;
    const int base0 = blk * 32;
    if (base0 >= n_nodes) return;

    const int col  = wave * 16 + m;      // B/C/D column this lane owns
    // Ragged tail: alias tile 1 onto tile 0 (compute discarded, store skipped)
    const bool has2 = (base0 + 16) < n_nodes;
    const int base1 = has2 ? (base0 + 16) : base0;

    const float* xrow0 = X + (size_t)(base0 + m) * F_IN;
    const float* xrow1 = X + (size_t)(base1 + m) * F_IN;
    const v8u*   bbase = (const v8u*)Bpack + (size_t)wave * 32 + lane;

    v8f acc0 = {};
    v8f acc1 = {};

    #pragma unroll
    for (int kk_i = 0; kk_i < 8; ++kk_i) {
        v16bf a0 = make_afrag(xrow0, kk_i, g);
        v16bf a1 = make_afrag(xrow1, kk_i, g);
        v8u bu = bbase[(size_t)kk_i * 256];
        v16bf b;
        __builtin_memcpy(&b, &bu, sizeof(b));
        acc0 = __builtin_amdgcn_wmma_f32_16x16x32_bf16(
            false, a0, false, b, (short)0, acc0, false, false);
        acc1 = __builtin_amdgcn_wmma_f32_16x16x32_bf16(
            false, a1, false, b, (short)0, acc1, false, false);
    }

    // D layout: lane owns column `col`; VGPR v holds row (base + v + 8g)
    #pragma unroll
    for (int v = 0; v < 8; ++v) {
        S[(size_t)(base0 + v + 8 * g) * F_OUT + col] = acc0[v];
    }
    if (has2) {
        #pragma unroll
        for (int v = 0; v < 8; ++v) {
            S[(size_t)(base1 + v + 8 * g) * F_OUT + col] = acc1[v];
        }
    }
}

// out[r] += val * support[c]; 32 lanes per edge, float4 per lane.
__global__ __launch_bounds__(256) void scatter_kernel(
    const int* __restrict__ erow, const int* __restrict__ ecol,
    const float* __restrict__ eval, const float* __restrict__ S,
    float* __restrict__ out, int n_edges) {
    int t = blockIdx.x * 256 + threadIdx.x;
    int e = t >> 5;
    if (e >= n_edges) return;
    int lane = t & 31;
    int r = erow[e];
    int c = ecol[e];
    float v = eval[e];
    float4 msg = *((const float4*)(S + (size_t)c * F_OUT) + lane);
    float* op = out + (size_t)r * F_OUT + lane * 4;
    unsafeAtomicAdd(op + 0, v * msg.x);   // global_atomic_add_f32, resolves in 192MB L2
    unsafeAtomicAdd(op + 1, v * msg.y);
    unsafeAtomicAdd(op + 2, v * msg.z);
    unsafeAtomicAdd(op + 3, v * msg.w);
}

extern "C" void kernel_launch(void* const* d_in, const int* in_sizes, int n_in,
                              void* d_out, int out_size, void* d_ws, size_t ws_size,
                              hipStream_t stream) {
    const float* x    = (const float*)d_in[0];
    const int*   erow = (const int*)d_in[1];
    const int*   ecol = (const int*)d_in[2];
    const float* eval = (const float*)d_in[3];
    const float* wgt  = (const float*)d_in[4];
    const float* bias = (const float*)d_in[5];
    float*       out  = (float*)d_out;

    const int n_nodes = in_sizes[0] / F_IN;
    const int n_edges = in_sizes[1];

    float*    S     = (float*)d_ws;                              // n_nodes*128 f32
    unsigned* Bpack = (unsigned*)(S + (size_t)n_nodes * F_OUT);  // 64 KB packed W frags

    // 1) out = bias (broadcast); fully re-initializes d_out each call.
    {
        int total  = n_nodes * F_OUT;
        int blocks = (total + 255) / 256;
        init_bias_kernel<<<blocks, 256, 0, stream>>>(bias, out, total);
    }
    // 2) pack W into WMMA B-fragment layout (tiny; L2-resident afterwards)
    pack_w_kernel<<<8, 256, 0, stream>>>(wgt, Bpack);
    // 3) S = X @ W via bf16 WMMA, 2 row tiles per wave
    {
        int blocks = (n_nodes + 31) / 32;
        gemm_xw_kernel<<<blocks, 256, 0, stream>>>(x, Bpack, S, n_nodes);
    }
    // 4) out[r] += val * S[c]
    {
        long long threads = (long long)n_edges * 32;
        int blocks = (int)((threads + 255) / 256);
        scatter_kernel<<<blocks, 256, 0, stream>>>(erow, ecol, eval, S, out, n_edges);
    }
}
